// MultimodalStateSpaceNetworks_60241211294439
// MI455X (gfx1250) — matrix operand order repaired and checked by reading, not compiled, not run on hardware
//
#include <hip/hip_runtime.h>

// Fused Multimodal Mamba pipeline for gfx1250 (MI455X, wave32, WMMA).
// One block = 4 batch elements (64 token rows), fully LDS-resident (~152 KB).
// GEMMs use v_wmma_f32_16x16x32_f16 (fp32 accum); scan is register-resident.

typedef _Float16 v16h __attribute__((ext_vector_type(16)));
typedef _Float16 h8   __attribute__((ext_vector_type(8)));
typedef float    v8f  __attribute__((ext_vector_type(8)));

#define BT      4                 // batch elements per block
#define ROWS    (BT * 16)         // 64 token rows per block
#define THREADS 256               // 8 waves
#define NWAVES  (THREADS / 32)

__device__ __forceinline__ float fast_sigmoid(float x) {
  return 1.f / (1.f + __expf(-x));
}

// A-matrix fragment (16x32 f16), ISA 7.12.2 layout:
// lane l: M = l&15 ; elem j: K = ((j>>3)<<4) + ((l>>4)<<3) + (j&7)
__device__ __forceinline__ v16h load_a_frag(const _Float16* base, int row0,
                                            int k0, int stride) {
  const int l = threadIdx.x & 31;
  const _Float16* p = base + (row0 + (l & 15)) * stride + k0 + ((l >> 4) << 3);
  h8 lo = *(const h8*)(p);
  h8 hi = *(const h8*)(p + 16);
  v16h v;
#pragma unroll
  for (int j = 0; j < 8; ++j) { v[j] = lo[j]; v[j + 8] = hi[j]; }
  return v;
}

// B-matrix fragment (32x16 f16) from weight stored [n][k]:
// lane l: N = l&15 ; elem j: K = ((j>>3)<<4) + ((l>>4)<<3) + (j&7)
__device__ __forceinline__ v16h load_b_frag(const _Float16* wbase, int n0,
                                            int k0, int stride) {
  const int l = threadIdx.x & 31;
  const _Float16* p = wbase + (n0 + (l & 15)) * stride + k0 + ((l >> 4) << 3);
  h8 lo = *(const h8*)(p);
  h8 hi = *(const h8*)(p + 16);
  v16h v;
#pragma unroll
  for (int j = 0; j < 8; ++j) { v[j] = lo[j]; v[j + 8] = hi[j]; }
  return v;
}

// C/D 16x16 f32: VGPR r -> M = r + ((l>=16)?8:0), N = l&15
__device__ __forceinline__ void store_d_frag(float* out, int row0, int col0,
                                             int stride, v8f d) {
  const int l = threadIdx.x & 31;
  float* p = out + (row0 + ((l >> 4) << 3)) * stride + col0 + (l & 15);
#pragma unroll
  for (int r = 0; r < 8; ++r) p[r * stride] = d[r];
}

__device__ __forceinline__ v8f wmma16x16x32(v16h a, v16h b, v8f c) {
  return __builtin_amdgcn_wmma_f32_16x16x32_f16(false, a, false, b, (short)0, c,
                                                false, false);
}

// ---------------------------------------------------------------------------
// One Mamba block on a 64-row token tile resident in LDS.
// ---------------------------------------------------------------------------
__device__ __forceinline__ void mamba_block(
    const float* h_in, float* h_out,                      // LDS 64x32 each
    _Float16* s_tok, float* s_xz, _Float16* s_xc,         // LDS activations
    float* s_dtbc, _Float16* s_yg,                        // LDS activations
    _Float16* s_win, _Float16* s_wbc, _Float16* s_wout,   // LDS f16 weights
    const float* g_inw, const float* g_convw, const float* g_convb,
    const float* g_xpw, const float* g_dtw, const float* g_dtb,
    const float* g_alog, const float* g_D, const float* g_outw) {
  const int tid = threadIdx.x;
  const int wave = tid >> 5;

  // ---- stage weights (f16, [out][k]) and tokens --------------------------
#pragma unroll
  for (int i = 0; i < 16; ++i) {                 // in_proj_w: 128x32 = 4096
    int idx = tid + i * THREADS;
    s_win[idx] = (_Float16)g_inw[idx];
  }
#pragma unroll
  for (int i = 0; i < 8; ++i) {                  // tokens: 64x32 = 2048
    int idx = tid + i * THREADS;
    s_tok[idx] = (_Float16)h_in[idx];
  }
#pragma unroll
  for (int i = 0; i < 48; ++i) {                 // dt_eff | Bproj | Cproj
    int idx = tid + i * THREADS;                 // 192*64 = 12288
    int n = idx >> 6, k = idx & 63;
    float v;
    if (n < 64)                                  // dt_proj_w @ x_proj_w[0:2]
      v = g_dtw[n * 2 + 0] * g_xpw[k] + g_dtw[n * 2 + 1] * g_xpw[64 + k];
    else if (n < 128)
      v = g_xpw[(2 + (n - 64)) * 64 + k];        // B projection rows
    else
      v = g_xpw[(66 + (n - 128)) * 64 + k];      // C projection rows
    s_wbc[idx] = (_Float16)v;
  }
#pragma unroll
  for (int i = 0; i < 8; ++i) {                  // out_proj_w: 32x64 = 2048
    int idx = tid + i * THREADS;
    s_wout[idx] = (_Float16)g_outw[idx];
  }
  __syncthreads();

  // ---- in_proj: (64x32) @ (32 -> 128), K=32 = one WMMA -------------------
  // 32 tile jobs, fully unrolled -> no loop control, EXEC stays all-ones.
#pragma unroll
  for (int i = 0; i < 32 / NWAVES; ++i) {
    int job = wave + i * NWAVES;
    int mt = job >> 3, nt = job & 7;
    v16h a = load_a_frag(s_tok, mt * 16, 0, 32);
    v16h b = load_b_frag(s_win, nt * 16, 0, 32);
    v8f acc = {};
    acc = wmma16x16x32(a, b, acc);
    store_d_frag(s_xz, mt * 16, nt * 16, 128, acc);
  }
  __syncthreads();

  // ---- causal depthwise conv (k=4) + SiLU -> xc (f16), branchless --------
#pragma unroll
  for (int i = 0; i < 16; ++i) {                 // 64x64 = 4096
    int idx = tid + i * THREADS;
    int row = idx >> 6, d = idx & 63;
    int t = row & 15;
    const float4 cw = *(const float4*)(g_convw + d * 4);
    const float cwk[4] = {cw.x, cw.y, cw.z, cw.w};
    float acc = g_convb[d];
#pragma unroll
    for (int k = 0; k < 4; ++k) {
      int tt = t - 3 + k;
      int rr = (tt >= 0) ? (row - 3 + k) : row;  // always-valid LDS address
      float xv = s_xz[rr * 128 + d];
      xv = (tt >= 0) ? xv : 0.f;                 // v_cndmask, no branch
      acc += xv * cwk[k];
    }
    acc = acc * fast_sigmoid(acc);
    s_xc[idx] = (_Float16)acc;
  }
  __syncthreads();

  // ---- dt/B/C projection: (64x64) @ (64 -> 192), K=64 = 2 WMMAs ----------
#pragma unroll
  for (int i = 0; i < 48 / NWAVES; ++i) {
    int job = wave + i * NWAVES;
    int mt = job / 12, nt = job % 12;
    v8f acc = {};
    acc = wmma16x16x32(load_a_frag(s_xc, mt * 16, 0, 64),
                       load_b_frag(s_wbc, nt * 16, 0, 64), acc);
    acc = wmma16x16x32(load_a_frag(s_xc, mt * 16, 32, 64),
                       load_b_frag(s_wbc, nt * 16, 32, 64), acc);
    store_d_frag(s_dtbc, mt * 16, nt * 16, 192, acc);
  }
  __syncthreads();

  // ---- dt = softplus(dt_lin + dt_bias) -----------------------------------
#pragma unroll
  for (int i = 0; i < 16; ++i) {                 // 64x64
    int idx = tid + i * THREADS;
    int row = idx >> 6, d = idx & 63;
    float v = s_dtbc[row * 192 + d] + g_dtb[d];
    v = (v > 20.f) ? v : __logf(1.f + __expf(v));
    s_dtbc[row * 192 + d] = v;
  }
  __syncthreads();

  // ---- selective scan: 2 waves per batch element, lane owns one d row ----
  {
    const int b = wave >> 1;
    const int d = ((wave & 1) << 5) + (tid & 31);
    float areg[64], hreg[64];
    const float4* Ap = (const float4*)(g_alog + d * 64);
#pragma unroll
    for (int n4 = 0; n4 < 16; ++n4) {
      float4 av = Ap[n4];
      areg[n4 * 4 + 0] = -__expf(av.x);
      areg[n4 * 4 + 1] = -__expf(av.y);
      areg[n4 * 4 + 2] = -__expf(av.z);
      areg[n4 * 4 + 3] = -__expf(av.w);
    }
#pragma unroll
    for (int n = 0; n < 64; ++n) hreg[n] = 0.f;
    const float Dd = g_D[d];

    for (int t = 0; t < 16; ++t) {
      const int row = b * 16 + t;
      const float dt_d = s_dtbc[row * 192 + d];
      const float x_d = (float)s_xc[row * 64 + d];
      const float scale = dt_d * x_d;
      const float4* Bp = (const float4*)(s_dtbc + row * 192 + 64);
      const float4* Cp = (const float4*)(s_dtbc + row * 192 + 128);
      float y = 0.f;
#pragma unroll
      for (int n4 = 0; n4 < 16; ++n4) {
        float4 bv = Bp[n4];   // uniform across lanes -> LDS broadcast
        float4 cv = Cp[n4];
        float bb[4] = {bv.x, bv.y, bv.z, bv.w};
        float cc[4] = {cv.x, cv.y, cv.z, cv.w};
#pragma unroll
        for (int i = 0; i < 4; ++i) {
          int n = n4 * 4 + i;
          float dA = __expf(dt_d * areg[n]);   // arg <= 0, no overflow
          hreg[n] = hreg[n] * dA + scale * bb[i];
          y += hreg[n] * cc[i];
        }
      }
      float z = s_xz[row * 128 + 64 + d];
      float yg = (y + x_d * Dd) * (z * fast_sigmoid(z));
      s_yg[row * 64 + d] = (_Float16)yg;
    }
  }
  __syncthreads();

  // ---- out_proj: (64x64) @ (64 -> 32), K=64 = 2 WMMAs --------------------
  {
    int job = wave;                              // exactly 8 jobs, 8 waves
    int mt = job >> 1, nt = job & 1;
    v8f acc = {};
    acc = wmma16x16x32(load_a_frag(s_yg, mt * 16, 0, 64),
                       load_b_frag(s_wout, nt * 16, 0, 64), acc);
    acc = wmma16x16x32(load_a_frag(s_yg, mt * 16, 32, 64),
                       load_b_frag(s_wout, nt * 16, 32, 64), acc);
    store_d_frag(h_out, mt * 16, nt * 16, 32, acc);
  }
  __syncthreads();
}

// ---------------------------------------------------------------------------
__global__ __launch_bounds__(THREADS, 1) void mssn_fused_kernel(
    const float* __restrict__ X, const float* __restrict__ embW,
    const float* __restrict__ embB,
    const float* m1_inw, const float* m1_convw, const float* m1_convb,
    const float* m1_xpw, const float* m1_dtw, const float* m1_dtb,
    const float* m1_alog, const float* m1_D, const float* m1_outw,
    const float* m2_inw, const float* m2_convw, const float* m2_convb,
    const float* m2_xpw, const float* m2_dtw, const float* m2_dtb,
    const float* m2_alog, const float* m2_D, const float* m2_outw,
    float* __restrict__ out) {
  __shared__ __align__(16) float    s_h0[ROWS * 32];      //  8 KB
  __shared__ __align__(16) float    s_h1[ROWS * 32];      //  8 KB
  __shared__ __align__(16) _Float16 s_tok[ROWS * 32];     //  4 KB
  __shared__ __align__(16) float    s_xz[ROWS * 128];     // 32 KB
  __shared__ __align__(16) _Float16 s_xc[ROWS * 64];      //  8 KB
  __shared__ __align__(16) float    s_dtbc[ROWS * 192];   // 48 KB
  __shared__ __align__(16) _Float16 s_yg[ROWS * 64];      //  8 KB
  __shared__ __align__(16) _Float16 s_win[128 * 32];      //  8 KB
  __shared__ __align__(16) _Float16 s_wbc[192 * 64];      // 24 KB
  __shared__ __align__(16) _Float16 s_wout[32 * 64];      //  4 KB

  const int tid = threadIdx.x;

  // ---- grouped embedding: h0[b,g,d] = embW[g,d,:] . X[b, g*8:...] + b ----
#pragma unroll
  for (int i = 0; i < 8; ++i) {                  // 64x32 = 2048 outputs
    int idx = tid + i * THREADS;
    int dm = idx & 31;
    int g = (idx >> 5) & 15;
    int b = idx >> 9;
    int gb = blockIdx.x * BT + b;
    const float* xr = X + gb * 128 + g * 8;
    const float* wr = embW + (g * 32 + dm) * 8;
    float acc = embB[g * 32 + dm];
#pragma unroll
    for (int f = 0; f < 8; ++f) acc += wr[f] * xr[f];
    s_h0[idx] = acc;
  }
  __syncthreads();

  mamba_block(s_h0, s_h1, s_tok, s_xz, s_xc, s_dtbc, s_yg, s_win, s_wbc, s_wout,
              m1_inw, m1_convw, m1_convb, m1_xpw, m1_dtw, m1_dtb, m1_alog, m1_D,
              m1_outw);
  mamba_block(s_h1, s_h0, s_tok, s_xz, s_xc, s_dtbc, s_yg, s_win, s_wbc, s_wout,
              m2_inw, m2_convw, m2_convb, m2_xpw, m2_dtw, m2_dtb, m2_alog, m2_D,
              m2_outw);

  // ---- residual: out = h2 + h1 -------------------------------------------
#pragma unroll
  for (int i = 0; i < 8; ++i) {
    int idx = tid + i * THREADS;
    out[blockIdx.x * (ROWS * 32) + idx] = s_h0[idx] + s_h1[idx];
  }
}

// ---------------------------------------------------------------------------
extern "C" void kernel_launch(void* const* d_in, const int* in_sizes, int n_in,
                              void* d_out, int out_size, void* d_ws,
                              size_t ws_size, hipStream_t stream) {
  (void)in_sizes; (void)n_in; (void)d_ws; (void)ws_size; (void)out_size;
  const float* X    = (const float*)d_in[0];
  const float* embW = (const float*)d_in[1];
  const float* embB = (const float*)d_in[2];
  const float* m1[9];
  const float* m2[9];
  for (int i = 0; i < 9; ++i) m1[i] = (const float*)d_in[3 + i];
  for (int i = 0; i < 9; ++i) m2[i] = (const float*)d_in[12 + i];
  float* out = (float*)d_out;

  dim3 grid(8192 / BT);   // 2048 blocks
  dim3 block(THREADS);
  mssn_fused_kernel<<<grid, block, 0, stream>>>(
      X, embW, embB,
      m1[0], m1[1], m1[2], m1[3], m1[4], m1[5], m1[6], m1[7], m1[8],
      m2[0], m2[1], m2[2], m2[3], m2[4], m2[5], m2[6], m2[7], m2[8],
      out);
}
